// UnsupervisedHebbianPlasticity_9448928051670
// MI455X (gfx1250) — matrix operand order repaired and checked by reading, not compiled, not run on hardware
//
#include <hip/hip_runtime.h>
#include <hip/hip_bf16.h>
#include <math.h>

// ---------------------------------------------------------------------------
// Problem constants
// ---------------------------------------------------------------------------
#define D_MODEL 2048
#define HDIM    256
#define GDIM    8
#define NPG     512
#define NTOK    16384      // B * N = 4 * 4096
#define BGRP    32         // B * G
#define RS      784        // raw row stride: [q(256) | k(256) | v(512..767) | d(768) | pad]

typedef __attribute__((ext_vector_type(16))) __bf16 v16bf;
typedef __attribute__((ext_vector_type(8)))  __bf16 v8bf;
typedef __attribute__((ext_vector_type(4)))  __bf16 v4bf;
typedef __attribute__((ext_vector_type(8)))  float  v8f;

union BF16x16 { v16bf v; struct { v8bf lo, hi; } h; };

// A-fragment (16x32 bf16, rows = output rows, K contiguous within rows).
// CDNA5 layout: lanes 0-15 hold K {0..7,16..23}, lanes 16-31 hold K {8..15,24..31}.
__device__ __forceinline__ v16bf ldsA(const __bf16* base, int stride, int row0,
                                      int k0, int lane) {
  int half = lane >> 4, r = lane & 15;
  const __bf16* p = base + (row0 + r) * stride + k0 + 8 * half;
  BF16x16 u;
  u.h.lo = *(const v8bf*)(p);
  u.h.hi = *(const v8bf*)(p + 16);
  return u.v;
}

// B-fragment (32x16 bf16) read from a TRANSPOSED staging buffer Bt where
// Bt[n][k] = B[k][n]. Lanes 0-15 hold K {0..7,16..23}, lanes 16-31 hold K {8..15,24..31}.
__device__ __forceinline__ v16bf ldsB(const __bf16* base, int stride, int col0,
                                      int k0, int lane) {
  int half = lane >> 4, n = lane & 15;
  const __bf16* p = base + (col0 + n) * stride + k0 + 8 * half;
  BF16x16 u;
  u.h.lo = *(const v8bf*)(p);
  u.h.hi = *(const v8bf*)(p + 16);
  return u.v;
}

__device__ __forceinline__ v8f wmma_bf16(v16bf a, v16bf b, v8f c) {
  return __builtin_amdgcn_wmma_f32_16x16x32_bf16(false, a, false, b,
                                                 (short)0, c, false, false);
}

// packed 4x bf16 store (emits v_cvt_pk_bf16_f32 x2 + ds_store_b64)
__device__ __forceinline__ void st4bf(__bf16* p, float4 v) {
  v4bf t;
  t[0] = (__bf16)v.x; t[1] = (__bf16)v.y; t[2] = (__bf16)v.z; t[3] = (__bf16)v.w;
  *(v4bf*)p = t;
}

// ---------------------------------------------------------------------------
// K1 staging helper (plain inlined function: no lambda capture frame, so
// nothing lands on the stack / scratch).
// ---------------------------------------------------------------------------
__device__ __forceinline__ void k1_stage(
    const float* __restrict__ x,  const float* __restrict__ Wq,
    const float* __restrict__ Wk, const float* __restrict__ Wv,
    const float* __restrict__ Wd, int tokBase, int colBase, int tid, int kb,
    __bf16* xsb, __bf16* wtb) {
  const int k0 = kb * 32;
  // x chunk: 256 rows x 32 k = 2048 float4
#pragma unroll
  for (int i = 0; i < 8; ++i) {
    int e4 = tid + 256 * i;
    int row = e4 >> 3, c4 = (e4 & 7) * 4;
    float4 a4 = *(const float4*)(x + (size_t)(tokBase + row) * D_MODEL + k0 + c4);
    st4bf(&xsb[row * 40 + c4], a4);
  }
  // W^T chunk: padded to 128 j x 32 k = 1024 float4 slots (shift/mask
  // indexing, no divergent guard; j>=112 rows are never read).
#pragma unroll
  for (int i = 0; i < 4; ++i) {
    int e4 = tid + 256 * i;
    int kk = e4 >> 5, j4 = (e4 & 31) * 4;
    int jg = colBase + j4;
    int kg = k0 + kk;
    float4 w4;
    if (jg < 768) {
      const float* base = (jg < 256) ? Wq : (jg < 512) ? Wk : Wv;
      w4 = *(const float4*)(base + (size_t)kg * HDIM + (jg & 255));
    } else {
      w4 = make_float4((jg == 768) ? Wd[kg] : 0.f, 0.f, 0.f, 0.f);
    }
    __bf16* p = &wtb[j4 * 40 + kk];
    p[0] = (__bf16)w4.x; p[40] = (__bf16)w4.y;
    p[80] = (__bf16)w4.z; p[120] = (__bf16)w4.w;
  }
}

// ---------------------------------------------------------------------------
// K1: raw[t][0..255]=x@Wq, [256..511]=x@Wk, [512..767]=x@Wv, [768]=x@Wd
// Grid: 64 token-blocks(256) x 7 col-blocks(112).  Double-buffered LDS.
// Per wave: 2 row-tiles x 7 col-tiles; B-fragments reused across row tiles.
// ---------------------------------------------------------------------------
__global__ __launch_bounds__(256, 1) void k_qkv_gemm(
    const float* __restrict__ x,  const float* __restrict__ Wq,
    const float* __restrict__ Wk, const float* __restrict__ Wv,
    const float* __restrict__ Wd, float* __restrict__ raw) {
  __shared__ __bf16 xs[2][256 * 40];   // 256 tokens x 32 k (stride 40)
  __shared__ __bf16 wt[2][128 * 40];   // 128 cols (transposed, rows>=112 pad) x 32 k

  const int tid = threadIdx.x, lane = tid & 31, wave = tid >> 5;
  const int tb = blockIdx.x / 7, cb = blockIdx.x % 7;
  const int tokBase = tb * 256, colBase = cb * 112;

  v8f acc[2][7];
#pragma unroll
  for (int rtl = 0; rtl < 2; ++rtl)
#pragma unroll
    for (int c = 0; c < 7; ++c)
#pragma unroll
      for (int e = 0; e < 8; ++e) acc[rtl][c][e] = 0.f;

  k1_stage(x, Wq, Wk, Wv, Wd, tokBase, colBase, tid, 0, xs[0], wt[0]);
  __syncthreads();
  for (int kb = 0; kb < D_MODEL / 32; ++kb) {
    const int cur = kb & 1;
    if (kb + 1 < D_MODEL / 32)
      k1_stage(x, Wq, Wk, Wv, Wd, tokBase, colBase, tid, kb + 1,
               xs[cur ^ 1], wt[cur ^ 1]);
    v16bf a0 = ldsA(xs[cur], 40, (2 * wave) * 16, 0, lane);
    v16bf a1 = ldsA(xs[cur], 40, (2 * wave) * 16 + 16, 0, lane);
#pragma unroll
    for (int c = 0; c < 7; ++c) {
      v16bf b = ldsB(wt[cur], 40, c * 16, 0, lane);
      acc[0][c] = wmma_bf16(a0, b, acc[0][c]);
      acc[1][c] = wmma_bf16(a1, b, acc[1][c]);
    }
    __syncthreads();
  }
  // epilogue: C layout lane L -> (row = r + 8*(L>>4), col = L&15)
  const int half = lane >> 4, n15 = lane & 15;
#pragma unroll
  for (int rtl = 0; rtl < 2; ++rtl)
#pragma unroll
    for (int c = 0; c < 7; ++c)
#pragma unroll
      for (int r = 0; r < 8; ++r) {
        int row = tokBase + (2 * wave + rtl) * 16 + r + 8 * half;
        int col = colBase + c * 16 + n15;
        raw[(size_t)row * RS + col] = acc[rtl][c][r];
      }
}

// ---------------------------------------------------------------------------
// K2: in-place row normalization. q,k: L2-normalize (+bias); v: LayerNorm.
// 32 tokens per block, 8 threads per token row.
// ---------------------------------------------------------------------------
__global__ __launch_bounds__(256) void k_normalize(
    float* __restrict__ raw, const float* __restrict__ bq,
    const float* __restrict__ bk, const float* __restrict__ bv,
    const float* __restrict__ lng, const float* __restrict__ lnb) {
  __shared__ float4 red[256];
  __shared__ float4 stat[32];
  const int tid = threadIdx.x, r = tid >> 3, l8 = tid & 7;
  float* rp = raw + (size_t)(blockIdx.x * 32 + r) * RS;

  float sq = 0.f, sk = 0.f, sv = 0.f, svv = 0.f;
#pragma unroll
  for (int i = 0; i < 8; ++i) {
    int c = (l8 + 8 * i) * 4;
    float4 Q = *(const float4*)(rp + c),       Bq = *(const float4*)(bq + c);
    float4 K = *(const float4*)(rp + 256 + c), Bk = *(const float4*)(bk + c);
    float4 V = *(const float4*)(rp + 512 + c), Bv = *(const float4*)(bv + c);
    float qx = Q.x + Bq.x, qy = Q.y + Bq.y, qz = Q.z + Bq.z, qw = Q.w + Bq.w;
    float kx = K.x + Bk.x, ky = K.y + Bk.y, kz = K.z + Bk.z, kw = K.w + Bk.w;
    float vx = V.x + Bv.x, vy = V.y + Bv.y, vz = V.z + Bv.z, vw = V.w + Bv.w;
    sq += qx * qx + qy * qy + qz * qz + qw * qw;
    sk += kx * kx + ky * ky + kz * kz + kw * kw;
    sv += vx + vy + vz + vw;
    svv += vx * vx + vy * vy + vz * vz + vw * vw;
  }
  red[tid] = make_float4(sq, sk, sv, svv);
  __syncthreads();
  if (l8 == 0) {
    float a = 0.f, b = 0.f, c = 0.f, d = 0.f;
#pragma unroll
    for (int j = 0; j < 8; ++j) {
      float4 v = red[r * 8 + j];
      a += v.x; b += v.y; c += v.z; d += v.w;
    }
    float qs = 1.f / fmaxf(sqrtf(a), 1e-5f);
    float ks = 1.f / fmaxf(sqrtf(b), 1e-5f);
    float mu = c * (1.f / 256.f);
    float var = d * (1.f / 256.f) - mu * mu;
    stat[r] = make_float4(qs, ks, mu, rsqrtf(var + 1e-5f));
  }
  __syncthreads();
  float4 st = stat[r];
#pragma unroll
  for (int i = 0; i < 8; ++i) {
    int c = (l8 + 8 * i) * 4;
    float4 Q = *(const float4*)(rp + c),       Bq = *(const float4*)(bq + c);
    float4 K = *(const float4*)(rp + 256 + c), Bk = *(const float4*)(bk + c);
    float4 V = *(const float4*)(rp + 512 + c), Bv = *(const float4*)(bv + c);
    float4 Gl = *(const float4*)(lng + c),     Bl = *(const float4*)(lnb + c);
    *(float4*)(rp + c) = make_float4((Q.x + Bq.x) * st.x, (Q.y + Bq.y) * st.x,
                                     (Q.z + Bq.z) * st.x, (Q.w + Bq.w) * st.x);
    *(float4*)(rp + 256 + c) = make_float4((K.x + Bk.x) * st.y, (K.y + Bk.y) * st.y,
                                           (K.z + Bk.z) * st.y, (K.w + Bk.w) * st.y);
    *(float4*)(rp + 512 + c) = make_float4(
        (V.x + Bv.x - st.z) * st.w * Gl.x + Bl.x,
        (V.y + Bv.y - st.z) * st.w * Gl.y + Bl.y,
        (V.z + Bv.z - st.z) * st.w * Gl.z + Bl.z,
        (V.w + Bv.w - st.z) * st.w * Gl.w + Bl.w);
  }
}

// ---------------------------------------------------------------------------
// K3: gamma[bg] = sigmoid(mean_n(d) + bd + timescale[g])
// ---------------------------------------------------------------------------
__global__ __launch_bounds__(256) void k_gamma(
    const float* __restrict__ raw, const float* __restrict__ bd,
    const float* __restrict__ ts, float* __restrict__ gamma) {
  __shared__ float red[256];
  const int bg = blockIdx.x, tid = threadIdx.x;
  red[tid] = raw[(size_t)(bg * 512 + tid) * RS + 768] +
             raw[(size_t)(bg * 512 + 256 + tid) * RS + 768];
  __syncthreads();
  for (int off = 128; off > 0; off >>= 1) {
    if (tid < off) red[tid] += red[tid + off];
    __syncthreads();
  }
  if (tid == 0) {
    float z = red[0] * (1.f / 512.f) + bd[0] + ts[bg & (GDIM - 1)];
    gamma[bg] = 1.f / (1.f + expf(-z));
  }
}

// ---------------------------------------------------------------------------
// K4/K6 shared: out_chunk[64,256] = A_chunk[64,256] @ M^T  (per bg group)
//   subtract=1: raw[:,oOff] -= result  (surprise = v - k M_prev^T, in place)
//   subtract=0: raw[:,oOff]  = result  (out_f = q M_next^T, in place over q)
// Grid: 32 groups x 8 token-chunks of 64.
// ---------------------------------------------------------------------------
__global__ __launch_bounds__(256, 1) void k_applyM(
    float* __restrict__ raw, const float* __restrict__ Msrc,
    int aOff, int oOff, int subtract) {
  __shared__ __bf16 as[64 * 264];   // 64 tokens x 256 feats
  __shared__ __bf16 ms[256 * 40];   // M slab: 256 rows(m) x 32 k  (B^T layout)
  const int tid = threadIdx.x, lane = tid & 31, wave = tid >> 5;
  const int bg = blockIdx.x >> 3, ch = blockIdx.x & 7;
  const int tBase = bg * 512 + ch * 64;
  const float* M = Msrc + (size_t)bg * (HDIM * HDIM);

#pragma unroll
  for (int i = 0; i < 16; ++i) {             // stage A once: 4096 float4
    int e = tid + 256 * i, row = e >> 6, c4 = (e & 63) * 4;
    float4 a4 = *(const float4*)(raw + (size_t)(tBase + row) * RS + aOff + c4);
    st4bf(&as[row * 264 + c4], a4);
  }
  v8f acc[4][2];
#pragma unroll
  for (int rt = 0; rt < 4; ++rt)
#pragma unroll
    for (int ct = 0; ct < 2; ++ct)
#pragma unroll
      for (int e = 0; e < 8; ++e) acc[rt][ct][e] = 0.f;

  for (int kb = 0; kb < 8; ++kb) {
    const int k0 = kb * 32;
#pragma unroll
    for (int i = 0; i < 8; ++i) {            // stage M slab: 2048 float4
      int e = tid + 256 * i, m = e >> 3, c4 = (e & 7) * 4;
      float4 m4 = *(const float4*)(M + m * HDIM + k0 + c4);
      st4bf(&ms[m * 40 + c4], m4);
    }
    __syncthreads();
    v16bf b0 = ldsB(ms, 40, (2 * wave) * 16, 0, lane);
    v16bf b1 = ldsB(ms, 40, (2 * wave) * 16 + 16, 0, lane);
#pragma unroll
    for (int rt = 0; rt < 4; ++rt) {
      v16bf a = ldsA(as, 264, rt * 16, k0, lane);
      acc[rt][0] = wmma_bf16(a, b0, acc[rt][0]);
      acc[rt][1] = wmma_bf16(a, b1, acc[rt][1]);
    }
    __syncthreads();
  }
  const int half = lane >> 4, n15 = lane & 15;
#pragma unroll
  for (int rt = 0; rt < 4; ++rt)
#pragma unroll
    for (int ct = 0; ct < 2; ++ct)
#pragma unroll
      for (int r = 0; r < 8; ++r) {
        int tok = tBase + rt * 16 + r + 8 * half;
        int feat = (2 * wave + ct) * 16 + n15;
        size_t idx = (size_t)tok * RS + oOff + feat;
        raw[idx] = subtract ? (raw[idx] - acc[rt][ct][r]) : acc[rt][ct][r];
      }
}

// ---------------------------------------------------------------------------
// K5 staging helper (plain inlined function; no lambda capture frame).
// ---------------------------------------------------------------------------
__device__ __forceinline__ void k5_stage(
    const float* __restrict__ raw, int bg, int mBase, int tid, int nb,
    __bf16* sTb, __bf16* kTb) {
  const int tBase = bg * 512 + nb * 32;
#pragma unroll
  for (int i = 0; i < 8; ++i) {            // sT: 64x32
    int e = tid + 256 * i, n = e >> 6, m = e & 63;
    sTb[m * 40 + n] = (__bf16)raw[(size_t)(tBase + n) * RS + 512 + mBase + m];
  }
#pragma unroll
  for (int i = 0; i < 32; ++i) {           // kT: 256x32
    int e = tid + 256 * i, n = e >> 8, h = e & 255;
    kTb[h * 40 + n] = (__bf16)raw[(size_t)(tBase + n) * RS + 256 + h];
  }
}

// ---------------------------------------------------------------------------
// K5: delta_M = surprise^T @ k / NPG ; M_next = clip(gamma*M_prev + delta)
// Grid: 32 groups x 4 slices of 64 m-rows. K = 512 tokens in chunks of 32.
// Both operands transposed into LDS at staging; double-buffered.
// ---------------------------------------------------------------------------
__global__ __launch_bounds__(256, 1) void k_deltaM(
    const float* __restrict__ raw, const float* __restrict__ Mprev,
    const float* __restrict__ gamma, float* __restrict__ Mnext) {
  __shared__ __bf16 sT[2][64 * 40];    // surprise^T: 64 m x 32 tokens
  __shared__ __bf16 kT[2][256 * 40];   // k^T: 256 h x 32 tokens
  const int tid = threadIdx.x, lane = tid & 31, wave = tid >> 5;
  const int bg = blockIdx.x >> 2, sl = blockIdx.x & 3;
  const int mBase = sl * 64;

  v8f acc[4][2];
#pragma unroll
  for (int rt = 0; rt < 4; ++rt)
#pragma unroll
    for (int ct = 0; ct < 2; ++ct)
#pragma unroll
      for (int e = 0; e < 8; ++e) acc[rt][ct][e] = 0.f;

  k5_stage(raw, bg, mBase, tid, 0, sT[0], kT[0]);
  __syncthreads();
  for (int nb = 0; nb < 16; ++nb) {
    const int cur = nb & 1;
    if (nb + 1 < 16)
      k5_stage(raw, bg, mBase, tid, nb + 1, sT[cur ^ 1], kT[cur ^ 1]);
    v16bf b0 = ldsB(kT[cur], 40, (2 * wave) * 16, 0, lane);
    v16bf b1 = ldsB(kT[cur], 40, (2 * wave) * 16 + 16, 0, lane);
#pragma unroll
    for (int rt = 0; rt < 4; ++rt) {
      v16bf a = ldsA(sT[cur], 40, rt * 16, 0, lane);
      acc[rt][0] = wmma_bf16(a, b0, acc[rt][0]);
      acc[rt][1] = wmma_bf16(a, b1, acc[rt][1]);
    }
    __syncthreads();
  }
  const float gm = gamma[bg];
  const int half = lane >> 4, n15 = lane & 15;
#pragma unroll
  for (int rt = 0; rt < 4; ++rt)
#pragma unroll
    for (int ct = 0; ct < 2; ++ct)
#pragma unroll
      for (int r = 0; r < 8; ++r) {
        int m = mBase + rt * 16 + r + 8 * half;
        int h = (2 * wave + ct) * 16 + n15;
        size_t idx = (size_t)bg * (HDIM * HDIM) + (size_t)m * HDIM + h;
        float v = gm * Mprev[idx] + acc[rt][ct][r] * (1.f / (float)NPG);
        Mnext[idx] = fminf(10.f, fmaxf(-10.f, v));
      }
}

// ---------------------------------------------------------------------------
// K7: out = out_f @ Wo + bo   [16384,256] x [256,2048]
// Grid: 256 token-blocks(64) x 8 j-blocks(256).
// ---------------------------------------------------------------------------
__global__ __launch_bounds__(256, 1) void k_outproj(
    const float* __restrict__ raw, const float* __restrict__ Wo,
    const float* __restrict__ bo, float* __restrict__ out) {
  __shared__ __bf16 as[64 * 264];   // out_f block: 64 tokens x 256 h
  __shared__ __bf16 wt[256 * 40];   // Wo^T slab: 256 j x 32 h
  const int tid = threadIdx.x, lane = tid & 31, wave = tid >> 5;
  const int tb = blockIdx.x >> 3, jb = blockIdx.x & 7;
  const int tBase = tb * 64, jBase = jb * 256;

#pragma unroll
  for (int i = 0; i < 16; ++i) {
    int e = tid + 256 * i, row = e >> 6, c4 = (e & 63) * 4;
    float4 a4 = *(const float4*)(raw + (size_t)(tBase + row) * RS + c4);
    st4bf(&as[row * 264 + c4], a4);
  }
  v8f acc[4][2];
#pragma unroll
  for (int rt = 0; rt < 4; ++rt)
#pragma unroll
    for (int ct = 0; ct < 2; ++ct)
#pragma unroll
      for (int e = 0; e < 8; ++e) acc[rt][ct][e] = 0.f;

  for (int kb = 0; kb < 8; ++kb) {
    const int k0 = kb * 32;
#pragma unroll
    for (int i = 0; i < 32; ++i) {           // wt[j][kk] = Wo[k0+kk][jBase+j]
      int e = tid + 256 * i, j = e & 255, kk = e >> 8;
      wt[j * 40 + kk] = (__bf16)Wo[(size_t)(k0 + kk) * D_MODEL + jBase + j];
    }
    __syncthreads();
    v16bf b0 = ldsB(wt, 40, (2 * wave) * 16, 0, lane);
    v16bf b1 = ldsB(wt, 40, (2 * wave) * 16 + 16, 0, lane);
#pragma unroll
    for (int rt = 0; rt < 4; ++rt) {
      v16bf a = ldsA(as, 264, rt * 16, k0, lane);
      acc[rt][0] = wmma_bf16(a, b0, acc[rt][0]);
      acc[rt][1] = wmma_bf16(a, b1, acc[rt][1]);
    }
    __syncthreads();
  }
  const int half = lane >> 4, n15 = lane & 15;
#pragma unroll
  for (int rt = 0; rt < 4; ++rt)
#pragma unroll
    for (int ct = 0; ct < 2; ++ct)
#pragma unroll
      for (int r = 0; r < 8; ++r) {
        int tok = tBase + rt * 16 + r + 8 * half;
        int j = jBase + (2 * wave + ct) * 16 + n15;
        out[(size_t)tok * D_MODEL + j] = acc[rt][ct][r] + bo[j];
      }
}

// ---------------------------------------------------------------------------
extern "C" void kernel_launch(void* const* d_in, const int* in_sizes, int n_in,
                              void* d_out, int out_size, void* d_ws,
                              size_t ws_size, hipStream_t stream) {
  (void)in_sizes; (void)n_in; (void)out_size; (void)ws_size;
  const float* x     = (const float*)d_in[0];
  const float* Mprev = (const float*)d_in[1];
  const float* Wq    = (const float*)d_in[2];
  const float* bq    = (const float*)d_in[3];
  const float* Wk    = (const float*)d_in[4];
  const float* bk    = (const float*)d_in[5];
  const float* Wv    = (const float*)d_in[6];
  const float* bv    = (const float*)d_in[7];
  const float* lng   = (const float*)d_in[8];
  const float* lnb   = (const float*)d_in[9];
  const float* Wo    = (const float*)d_in[10];
  const float* bo    = (const float*)d_in[11];
  const float* Wd    = (const float*)d_in[12];
  const float* bd    = (const float*)d_in[13];
  const float* ts    = (const float*)d_in[14];

  float* out   = (float*)d_out;                         // [16384, 2048]
  float* Mnext = out + (size_t)NTOK * D_MODEL;          // [32, 256, 256]
  float* raw   = (float*)d_ws;                          // [16384, 784]
  float* gamma = raw + (size_t)NTOK * RS;               // [32]

  k_qkv_gemm <<<64 * 7, 256, 0, stream>>>(x, Wq, Wk, Wv, Wd, raw);
  k_normalize<<<NTOK / 32, 256, 0, stream>>>(raw, bq, bk, bv, lng, lnb);
  k_gamma    <<<BGRP, 256, 0, stream>>>(raw, bd, ts, gamma);
  // surprise = v - k @ M_prev^T   (in place over v columns)
  k_applyM   <<<BGRP * 8, 256, 0, stream>>>(raw, Mprev, 256, 512, 1);
  // M_next = clip(gamma*M_prev + surprise^T k / NPG)
  k_deltaM   <<<BGRP * 4, 256, 0, stream>>>(raw, Mprev, gamma, Mnext);
  // out_f = q @ M_next^T   (in place over q columns)
  k_applyM   <<<BGRP * 8, 256, 0, stream>>>(raw, Mnext, 0, 0, 0);
  k_outproj  <<<(NTOK / 64) * 8, 256, 0, stream>>>(raw, Wo, bo, out);
}
